// SinkhornDistance_57501022159586
// MI455X (gfx1250) — compile-verified, matrix operand order checked by reading, not run on hardware
//
#include <hip/hip_runtime.h>

// ---------------------------------------------------------------------------
// Sinkhorn distance for n=m=4096, d=256 on gfx1250 (MI455X).
//   C = cdist(x,y); C /= mean(C); K = exp(-C/EPS)
//   2000x: u = a/(K v + stab); v = b/(K^T u + stab)
//   cost = sum(u K v * C)
// Output: [cost, u(4096), v(4096)]  (8193 f32)
// ---------------------------------------------------------------------------

typedef __bf16 bf16;
typedef __attribute__((ext_vector_type(16))) bf16  v16bf;
typedef __attribute__((ext_vector_type(8)))  bf16  v8bf;
typedef __attribute__((ext_vector_type(8)))  float v8f;

typedef __attribute__((ext_vector_type(4))) unsigned int u32x4;
typedef __attribute__((ext_vector_type(4))) int          i32x4;
typedef __attribute__((ext_vector_type(8))) int          i32x8;

static constexpr int   Ndim  = 4096;
static constexpr int   Ddim  = 256;
static constexpr float kEPS  = 1e-4f;
static constexpr float kSTAB = 1e-8f;
static constexpr int   kIters = 2000;
static constexpr int   kSinkBlocks = 128;   // persistent kernel; must be co-resident

// ------------------------- TDM 1-D global -> LDS copy ----------------------
// Builds a Tensor DMA Descriptor (D#) for a single row of `nelem` f32 values
// and issues TENSOR_LOAD_TO_LDS. Groups 2/3 unused (<=2-D tensor).
#if defined(__has_builtin) && __has_builtin(__builtin_amdgcn_tensor_load_to_lds)
#define USE_TDM 1
#else
#define USE_TDM 0
#endif

#if USE_TDM
__device__ __forceinline__ void tdm_copy_1d_f32(unsigned lds_off, const float* gsrc, int nelem) {
    unsigned long long ga = (unsigned long long)(__SIZE_TYPE__)gsrc;
    u32x4 g0;
    g0[0] = 1u;                                              // count=1, user mode
    g0[1] = lds_off;                                         // lds_addr (bytes)
    g0[2] = (unsigned)(ga & 0xFFFFFFFFu);                    // global_addr[31:0]
    g0[3] = (unsigned)((ga >> 32) & 0x01FFFFFFu)             // global_addr[56:32]
          | (2u << 30);                                      // type = 2 ("image")
    i32x8 g1;
    g1[0] = (2 << 16);                                       // data_size = 4 bytes
    g1[1] = (int)((unsigned)(nelem & 0xFFFF) << 16);         // tensor_dim0[15:0]
    g1[2] = (int)(((unsigned)nelem >> 16) | (1u << 16));     // tensor_dim0[31:16], tensor_dim1=1
    g1[3] = (int)((unsigned)(nelem & 0xFFFF) << 16);         // tile_dim0 = nelem
    g1[4] = 1;                                               // tile_dim1 = 1
    g1[5] = nelem;                                           // tensor_dim0_stride
    g1[6] = 0;
    g1[7] = 0;
    i32x4 z4 = {0, 0, 0, 0};
#if __has_include(<hip/amd_detail/amd_gfx1250_TDM.h>)
    i32x8 z8 = {0, 0, 0, 0, 0, 0, 0, 0};
    __builtin_amdgcn_tensor_load_to_lds(g0, g1, z4, z4, z8, 0);
#else
    __builtin_amdgcn_tensor_load_to_lds(g0, g1, z4, z4, 0);
#endif
}
#endif

// Stage `nelem` f32 from global into LDS, block-wide. TDM path: one wave
// issues the DMA and waits on TENSORcnt; __syncthreads releases the block.
__device__ __forceinline__ void stage_to_lds(float* sdst, const float* gsrc, int nelem) {
#if USE_TDM
    if (threadIdx.x == 0)
        tdm_copy_1d_f32((unsigned)(__SIZE_TYPE__)sdst, gsrc, nelem);
    __builtin_amdgcn_s_wait_tensorcnt(0);   // no-op for waves with TENSORcnt==0
    __syncthreads();
#else
    for (int i = threadIdx.x; i < nelem; i += 256) sdst[i] = gsrc[i];
    __syncthreads();
#endif
}

// -------------------------------- grid sync --------------------------------
__device__ __forceinline__ void grid_sync(unsigned* bar, unsigned nblocks) {
    __syncthreads();
    if (threadIdx.x == 0) {
        unsigned gen  = __hip_atomic_load(&bar[1], __ATOMIC_RELAXED, __HIP_MEMORY_SCOPE_AGENT);
        unsigned prev = __hip_atomic_fetch_add(&bar[0], 1u, __ATOMIC_ACQ_REL, __HIP_MEMORY_SCOPE_AGENT);
        if (prev + 1u == nblocks) {
            __hip_atomic_store(&bar[0], 0u, __ATOMIC_RELAXED, __HIP_MEMORY_SCOPE_AGENT);
            __hip_atomic_fetch_add(&bar[1], 1u, __ATOMIC_RELEASE, __HIP_MEMORY_SCOPE_AGENT);
        } else {
            while (__hip_atomic_load(&bar[1], __ATOMIC_ACQUIRE, __HIP_MEMORY_SCOPE_AGENT) == gen)
                __builtin_amdgcn_s_sleep(1);
        }
    }
    __syncthreads();
}

// ---------------------- prep: f32 -> bf16 + row norms ----------------------
__global__ __launch_bounds__(256) void prep_kernel(
    const float* __restrict__ x, const float* __restrict__ y,
    bf16* __restrict__ xb, bf16* __restrict__ yb,
    float* __restrict__ x2, float* __restrict__ y2)
{
    int gw   = (blockIdx.x * 256 + threadIdx.x) >> 5;
    int lane = threadIdx.x & 31;
    const float* src; bf16* dst; float* nrm; int r;
    if (gw < Ndim) { r = gw;        src = x + (size_t)r * Ddim; dst = xb + (size_t)r * Ddim; nrm = x2; }
    else           { r = gw - Ndim; src = y + (size_t)r * Ddim; dst = yb + (size_t)r * Ddim; nrm = y2; }
    float s = 0.f;
    #pragma unroll
    for (int j = lane; j < Ddim; j += 32) {
        float f = src[j];
        s += f * f;
        dst[j] = (bf16)f;
    }
    #pragma unroll
    for (int off = 16; off; off >>= 1) s += __shfl_xor(s, off, 32);
    if (lane == 0) nrm[r] = s;
}

// ------------------- GEMM + cdist epilogue via bf16 WMMA -------------------
// Each wave32 owns one 16x16 tile of C; 8 waves/block -> 8192 blocks.
__global__ __launch_bounds__(256) void gemm_cdist_kernel(
    const bf16* __restrict__ xb, const bf16* __restrict__ yb,
    const float* __restrict__ x2, const float* __restrict__ y2,
    float* __restrict__ C, float* __restrict__ Csum)
{
    int wid  = threadIdx.x >> 5;
    int lane = threadIdx.x & 31;
    int t    = blockIdx.x * 8 + wid;          // 0 .. 65535 tiles
    int tileM = (t >> 8) << 4;
    int tileN = (t & 255) << 4;
    int h = lane >> 4;
    int l = lane & 15;

    const bf16* arow = xb + (size_t)(tileM + l) * Ddim;
    const bf16* brow = yb + (size_t)(tileN + l) * Ddim;

    v8f acc = {0.f, 0.f, 0.f, 0.f, 0.f, 0.f, 0.f, 0.f};
    #pragma unroll
    for (int k0 = 0; k0 < Ddim; k0 += 32) {
        v8bf a0 = *(const v8bf*)(arow + k0 + 8 * h);
        v8bf a1 = *(const v8bf*)(arow + k0 + 16 + 8 * h);
        v8bf b0 = *(const v8bf*)(brow + k0 + 8 * h);
        v8bf b1 = *(const v8bf*)(brow + k0 + 16 + 8 * h);
        v16bf a, b;
        #pragma unroll
        for (int e = 0; e < 8; ++e) {
            a[e] = a0[e]; a[8 + e] = a1[e];
            b[e] = b0[e]; b[8 + e] = b1[e];
        }
        acc = __builtin_amdgcn_wmma_f32_16x16x32_bf16(
            /*neg_a=*/false, a, /*neg_b=*/false, b,
            /*c_mod=*/(short)0, acc, /*reuse_a=*/false, /*reuse_b=*/false);
    }

    int   n = tileN + l;
    float yn = y2[n];
    float local = 0.f;
    #pragma unroll
    for (int r = 0; r < 8; ++r) {
        int   m = tileM + 8 * h + r;
        float sq = x2[m] + yn - 2.0f * acc[r];
        float c  = sqrtf(fmaxf(sq, 0.0f));
        C[(size_t)m * Ndim + n] = c;
        local += c;
    }
    #pragma unroll
    for (int off = 16; off; off >>= 1) local += __shfl_xor(local, off, 32);
    if (lane == 0) atomicAdd(Csum, local);
}

// -------- gibbs: K = exp(-C/(mean*EPS)) in bf16, plus K^T (LDS transpose) --
__global__ __launch_bounds__(256) void gibbs_kernel(
    const float* __restrict__ C, const float* __restrict__ Csum,
    bf16* __restrict__ K, bf16* __restrict__ KT)
{
    __shared__ float tile[32][33];
    int bx = blockIdx.x & 127;
    int by = blockIdx.x >> 7;
    int tx = threadIdx.x & 31;
    int ty = threadIdx.x >> 5;

    float mean = Csum[0] * (1.0f / ((float)Ndim * (float)Ndim));
    float scal = -1.0f / (mean * kEPS);

    int col = bx * 32 + tx;
    #pragma unroll
    for (int rr = 0; rr < 32; rr += 8) {
        int row = by * 32 + ty + rr;
        float c = C[(size_t)row * Ndim + col];
        float k = __expf(c * scal);
        K[(size_t)row * Ndim + col] = (bf16)k;
        tile[ty + rr][tx] = k;
    }
    __syncthreads();
    #pragma unroll
    for (int rr = 0; rr < 32; rr += 8) {
        int trow = bx * 32 + ty + rr;
        int tcol = by * 32 + tx;
        KT[(size_t)trow * Ndim + tcol] = (bf16)tile[tx][ty + rr];
    }
}

// ----------------- persistent Sinkhorn fixed-point iteration ---------------
__global__ __launch_bounds__(256) void sinkhorn_kernel(
    const bf16* __restrict__ K, const bf16* __restrict__ KT,
    float* __restrict__ u, float* __restrict__ v, unsigned* __restrict__ bar)
{
    __shared__ float sv[Ndim];                 // 16 KB of the 320 KB/WGP LDS
    const int tid    = threadIdx.x;
    const int lane   = tid & 31;
    const int gthr   = blockIdx.x * 256 + tid;
    const int nthr   = gridDim.x * 256;
    const int gwave  = gthr >> 5;
    const int nwaves = nthr >> 5;
    const float a    = 1.0f / (float)Ndim;     // a == b == 1/n

    for (int i = gthr; i < Ndim; i += nthr) v[i] = 0.f;   // v0 = 0
    __threadfence();
    grid_sync(bar, gridDim.x);

    for (int it = 0; it < kIters; ++it) {
        // ---- phase A: u = a / (K v + stab) ----
        stage_to_lds(sv, v, Ndim);             // TDM broadcast of v into LDS
        for (int r = gwave; r < Ndim; r += nwaves) {
            const bf16* row = K + (size_t)r * Ndim;
            if (r + nwaves < Ndim)
                __builtin_prefetch((const void*)(K + (size_t)(r + nwaves) * Ndim), 0, 1);
            float s = 0.f;
            #pragma unroll 4
            for (int jj = lane * 8; jj < Ndim; jj += 32 * 8) {
                v8bf kk = *(const v8bf*)(row + jj);
                #pragma unroll
                for (int e = 0; e < 8; ++e) s += (float)kk[e] * sv[jj + e];
            }
            #pragma unroll
            for (int off = 16; off; off >>= 1) s += __shfl_xor(s, off, 32);
            if (lane == 0) u[r] = a / (s + kSTAB);
        }
        __threadfence();
        grid_sync(bar, gridDim.x);

        // ---- phase B: v = b / (K^T u + stab) ----
        stage_to_lds(sv, u, Ndim);             // TDM broadcast of u into LDS
        for (int r = gwave; r < Ndim; r += nwaves) {
            const bf16* row = KT + (size_t)r * Ndim;
            if (r + nwaves < Ndim)
                __builtin_prefetch((const void*)(KT + (size_t)(r + nwaves) * Ndim), 0, 1);
            float s = 0.f;
            #pragma unroll 4
            for (int jj = lane * 8; jj < Ndim; jj += 32 * 8) {
                v8bf kk = *(const v8bf*)(row + jj);
                #pragma unroll
                for (int e = 0; e < 8; ++e) s += (float)kk[e] * sv[jj + e];
            }
            #pragma unroll
            for (int off = 16; off; off >>= 1) s += __shfl_xor(s, off, 32);
            if (lane == 0) v[r] = a / (s + kSTAB);
        }
        __threadfence();
        grid_sync(bar, gridDim.x);
    }
}

// ------------------------- cost = sum(u K v * C/mean) ----------------------
__global__ __launch_bounds__(256) void cost_kernel(
    const float* __restrict__ C, const bf16* __restrict__ K,
    const float* __restrict__ u, const float* __restrict__ v,
    const float* __restrict__ Csum, float* __restrict__ out)
{
    __shared__ float red[8];
    float invMean = ((float)Ndim * (float)Ndim) / Csum[0];
    size_t total  = (size_t)Ndim * Ndim;
    size_t stride = (size_t)gridDim.x * 256;
    float s = 0.f;
    for (size_t idx = (size_t)blockIdx.x * 256 + threadIdx.x; idx < total; idx += stride) {
        int r = (int)(idx >> 12);
        int c = (int)(idx & (Ndim - 1));
        s += u[r] * (float)K[idx] * v[c] * (C[idx] * invMean);
    }
    #pragma unroll
    for (int off = 16; off; off >>= 1) s += __shfl_xor(s, off, 32);
    if ((threadIdx.x & 31) == 0) red[threadIdx.x >> 5] = s;
    __syncthreads();
    if (threadIdx.x < 8) {
        float t = red[threadIdx.x];
        #pragma unroll
        for (int off = 4; off; off >>= 1) t += __shfl_xor(t, off, 32);
        if (threadIdx.x == 0) atomicAdd(out, t);
    }
}

__global__ __launch_bounds__(256) void finalize_kernel(
    const float* __restrict__ u, const float* __restrict__ v, float* __restrict__ out)
{
    int i = blockIdx.x * 256 + threadIdx.x;
    if (i < Ndim) {
        out[1 + i]        = u[i];
        out[1 + Ndim + i] = v[i];
    }
}

// --------------------------------- launch ----------------------------------
extern "C" void kernel_launch(void* const* d_in, const int* in_sizes, int n_in,
                              void* d_out, int out_size, void* d_ws, size_t ws_size,
                              hipStream_t stream) {
    const float* x = (const float*)d_in[0];
    const float* y = (const float*)d_in[1];
    float* out = (float*)d_out;

    char* p = (char*)d_ws;
    auto take = [&](size_t bytes) -> char* {
        char* r = p;
        p += (bytes + 255) & ~(size_t)255;
        return r;
    };
    bf16*     xb   = (bf16*)    take((size_t)Ndim * Ddim * sizeof(bf16));   //  2 MB
    bf16*     yb   = (bf16*)    take((size_t)Ndim * Ddim * sizeof(bf16));   //  2 MB
    float*    x2   = (float*)   take(Ndim * sizeof(float));
    float*    y2   = (float*)   take(Ndim * sizeof(float));
    float*    u    = (float*)   take(Ndim * sizeof(float));
    float*    v    = (float*)   take(Ndim * sizeof(float));
    float*    Csum = (float*)   take(256);
    unsigned* bar  = (unsigned*)take(256);
    float*    C    = (float*)   take((size_t)Ndim * Ndim * sizeof(float));  // 64 MB
    bf16*     K    = (bf16*)    take((size_t)Ndim * Ndim * sizeof(bf16));   // 32 MB
    bf16*     KT   = (bf16*)    take((size_t)Ndim * Ndim * sizeof(bf16));   // 32 MB

    hipMemsetAsync(Csum, 0, sizeof(float), stream);
    hipMemsetAsync(bar,  0, 2 * sizeof(unsigned), stream);
    hipMemsetAsync(out,  0, sizeof(float), stream);

    prep_kernel<<<1024, 256, 0, stream>>>(x, y, xb, yb, x2, y2);
    gemm_cdist_kernel<<<8192, 256, 0, stream>>>(xb, yb, x2, y2, C, Csum);
    gibbs_kernel<<<128 * 128, 256, 0, stream>>>(C, Csum, K, KT);
    sinkhorn_kernel<<<kSinkBlocks, 256, 0, stream>>>(K, KT, u, v, bar);
    cost_kernel<<<2048, 256, 0, stream>>>(C, K, u, v, Csum, out);
    finalize_kernel<<<(Ndim + 255) / 256, 256, 0, stream>>>(u, v, out);
}
